// SinkhornDistance_24086176596691
// MI455X (gfx1250) — matrix operand order repaired and checked by reading, not compiled, not run on hardware
//
#include <hip/hip_runtime.h>
#include <hip/hip_bf16.h>

typedef __attribute__((ext_vector_type(16))) _Float16 v16h;
typedef __attribute__((ext_vector_type(8)))  _Float16 v8h;
typedef __attribute__((ext_vector_type(8)))  float    v8f;

constexpr int   N       = 4096;
constexpr float EPS     = 0.1f;
constexpr float INV_EPS = 10.0f;
constexpr int   ITERS   = 50;
constexpr int   NSPLIT  = 16;   // row/col splits per GEMV pass (occupancy)

// ---------------------------------------------------------------------------
// Build C (f32, into d_out) and K = exp(-C/eps) (f16, resident in L2)
// ---------------------------------------------------------------------------
__global__ void build_CK(const float* __restrict__ x, const float* __restrict__ y,
                         float* __restrict__ C, _Float16* __restrict__ Kh) {
    int gid = blockIdx.x * blockDim.x + threadIdx.x;       // 0 .. N*N-1
    int i = gid >> 12, j = gid & (N - 1);
    float c = fabsf(x[2*i]   - y[2*j])
            + fabsf(x[2*i+1] - y[2*j+1]);                  // P = 1 cost
    C[gid]  = c;
    Kh[gid] = (_Float16)__expf(-c * INV_EPS);
}

__global__ void init_b(_Float16* __restrict__ b_h, float* __restrict__ msl) {
    int t = blockIdx.x * blockDim.x + threadIdx.x;
    if (t < N)  b_h[t] = (_Float16)1.0f;                   // v = 0 -> b = 1
    if (t == 0) msl[0] = 0.0f;                             // m_v = 0
}

// ---------------------------------------------------------------------------
// Row pass: partial S_i = sum_j K[i][j] * b[j]  via v_wmma_f32_16x16x32_f16
// A = 16x32 K tile, B = b chunk replicated over 16 columns.
// ---------------------------------------------------------------------------
__global__ void gemv_row(const _Float16* __restrict__ Kh,
                         const _Float16* __restrict__ b_h,
                         float* __restrict__ partials) {
    int wave = (blockIdx.x * blockDim.x + threadIdx.x) >> 5;   // 0..4095
    int lane = threadIdx.x & 31;
    int rg = wave >> 4;                // row group (16 rows)
    int s  = wave & (NSPLIT - 1);      // column split
    int i0 = rg * 16;
    int h = lane >> 4, l = lane & 15;

    v8f acc = {};
    const _Float16* rowp = Kh + (size_t)(i0 + l) * N + h * 16;
    for (int j0 = s * 256; j0 < s * 256 + 256; j0 += 32) {
        // A layout (f16 16x32): lane half h holds K = h*16..h*16+15
        v8h alo = *(const v8h*)(rowp + j0);
        v8h ahi = *(const v8h*)(rowp + j0 + 8);
        v16h A;
        for (int e = 0; e < 8; ++e) { A[e] = alo[e]; A[e + 8] = ahi[e]; }
        // B layout (f16 32x16): lane = column, halves e = K (h*16 + e); b replicated
        v16h B = *(const v16h*)(b_h + j0 + h * 16);
        acc = __builtin_amdgcn_wmma_f32_16x16x32_f16(
                  false, A, false, B, (short)0, acc, false, false);
    }
    // D: lanes 0-15 hold rows i0+0..7 in acc[0..7]; lanes 16-31 rows i0+8..15.
    if (l == 0)
        for (int g = 0; g < 8; ++g)
            partials[s * N + i0 + h * 8 + g] = acc[g];
}

// ---------------------------------------------------------------------------
// Col pass: partial T_j = sum_i a[i] * K[i][j]
// A = a chunk replicated over 16 rows, B = 32x16 K tile (column-wise loads).
// ---------------------------------------------------------------------------
__global__ void gemv_col(const _Float16* __restrict__ Kh,
                         const _Float16* __restrict__ a_h,
                         float* __restrict__ partials) {
    int wave = (blockIdx.x * blockDim.x + threadIdx.x) >> 5;
    int lane = threadIdx.x & 31;
    int cg = wave >> 4;                // column group (16 cols)
    int s  = wave & (NSPLIT - 1);      // row split
    int j0 = cg * 16;
    int h = lane >> 4, l = lane & 15;

    v8f acc = {};
    for (int i0 = s * 256; i0 < s * 256 + 256; i0 += 32) {
        v8h lo = *(const v8h*)(a_h + i0 + h * 16);
        v8h hi = *(const v8h*)(a_h + i0 + h * 16 + 8);
        v16h A;
        for (int e = 0; e < 8; ++e) { A[e] = lo[e]; A[e + 8] = hi[e]; }
        v16h B;
        const _Float16* bp = Kh + (size_t)(i0 + h * 16) * N + j0 + l;
        for (int e = 0; e < 16; ++e) B[e] = bp[(size_t)e * N];
        acc = __builtin_amdgcn_wmma_f32_16x16x32_f16(
                  false, A, false, B, (short)0, acc, false, false);
    }
    if (lane < 16) partials[s * N + j0 + l] = acc[0];      // replicated over M
}

// ---------------------------------------------------------------------------
// Combine partials -> new potential (u or v), its max, and scaled exp vector.
//   u_i = eps*log_w - m_other - eps*log(S_i);  a_i = exp((u_i - max u)/eps)
// Single block, fully deterministic.
// ---------------------------------------------------------------------------
__global__ void combine(const float* __restrict__ partials,
                        float* __restrict__ uv, _Float16* __restrict__ ab,
                        float* __restrict__ msl, int read_m, int write_m,
                        float eps_log_w) {
    __shared__ float red[1024];
    int t = threadIdx.x;                                   // blockDim = 1024
    float m_other = msl[read_m];
    float vals[4];
    float localmax = -1e30f;
    for (int k = 0; k < 4; ++k) {
        int r = t * 4 + k;
        float S = 0.f;
        for (int s2 = 0; s2 < NSPLIT; ++s2) S += partials[s2 * N + r];
        float val = eps_log_w - m_other - EPS * __logf(fmaxf(S, 1e-30f));
        uv[r] = val; vals[k] = val;
        localmax = fmaxf(localmax, val);
    }
    red[t] = localmax; __syncthreads();
    for (int st = 512; st > 0; st >>= 1) {
        if (t < st) red[t] = fmaxf(red[t], red[t + st]);
        __syncthreads();
    }
    float m = red[0];
    for (int k = 0; k < 4; ++k)
        ab[t * 4 + k] = (_Float16)__expf((vals[k] - m) * INV_EPS);
    if (t == 0) msl[write_m] = m;
}

// ---------------------------------------------------------------------------
// pi = exp((-C + u + v)/eps) from full-precision C; per-block cost partials.
// ---------------------------------------------------------------------------
__global__ void finalize(const float* __restrict__ C, const float* __restrict__ u,
                         const float* __restrict__ v, float* __restrict__ pi,
                         float* __restrict__ partials) {
    __shared__ float red[256];
    int gid = blockIdx.x * 256 + threadIdx.x;
    int i = gid >> 12, j = gid & (N - 1);
    float c = C[gid];
    float p = __expf((u[i] + v[j] - c) * INV_EPS);
    pi[gid] = p;
    red[threadIdx.x] = p * c; __syncthreads();
    for (int st = 128; st > 0; st >>= 1) {
        if (threadIdx.x < st) red[threadIdx.x] += red[threadIdx.x + st];
        __syncthreads();
    }
    if (threadIdx.x == 0) partials[blockIdx.x] = red[0];
}

__global__ void reduce_cost(const float* __restrict__ partials,
                            float* __restrict__ out) {
    __shared__ float red[1024];
    int t = threadIdx.x;
    float s = 0.f;
    for (int k = t; k < (N * N) / 256; k += 1024) s += partials[k];
    red[t] = s; __syncthreads();
    for (int st = 512; st > 0; st >>= 1) {
        if (t < st) red[t] += red[t + st];
        __syncthreads();
    }
    if (t == 0) out[0] = red[0];                           // cost^(1/P), P=1
}

// ---------------------------------------------------------------------------
extern "C" void kernel_launch(void* const* d_in, const int* in_sizes, int n_in,
                              void* d_out, int out_size, void* d_ws, size_t ws_size,
                              hipStream_t stream) {
    const float* x = (const float*)d_in[0];
    const float* y = (const float*)d_in[1];
    float* out = (float*)d_out;
    float* pi  = out + 1;
    float* C   = out + 1 + (size_t)N * N;
    // K (f16, 33.5 MB) parked inside the pi region (dead until finalize),
    // 256-byte aligned assuming d_out is allocation-aligned.
    _Float16* Kh = (_Float16*)((char*)d_out + 256);

    char* ws = (char*)d_ws;                                // ~310 KB used
    float*    partials = (float*)ws;                       // 65536 f32 (reused)
    float*    u   = (float*)(ws + 262144);
    float*    v   = (float*)(ws + 262144 + 16384);
    _Float16* a_h = (_Float16*)(ws + 262144 + 32768);
    _Float16* b_h = (_Float16*)(ws + 262144 + 32768 + 8192);
    float*    msl = (float*)(ws + 262144 + 32768 + 16384); // [0]=m_v, [1]=m_u

    const float eps_log_w = EPS * logf(1.0f / N + 1e-8f);  // log_mu == log_nu
    const int gemv_blocks = (N / 16) * NSPLIT * 32 / 256;  // 512 blocks, 8 waves ea.

    build_CK<<<N * N / 256, 256, 0, stream>>>(x, y, C, Kh);
    init_b<<<N / 256, 256, 0, stream>>>(b_h, msl);
    for (int it = 0; it < ITERS; ++it) {
        gemv_row<<<gemv_blocks, 256, 0, stream>>>(Kh, b_h, partials);
        combine<<<1, 1024, 0, stream>>>(partials, u, a_h, msl, 0, 1, eps_log_w);
        gemv_col<<<gemv_blocks, 256, 0, stream>>>(Kh, a_h, partials);
        combine<<<1, 1024, 0, stream>>>(partials, v, b_h, msl, 1, 0, eps_log_w);
    }
    finalize<<<N * N / 256, 256, 0, stream>>>(C, u, v, pi, partials);
    reduce_cost<<<1, 1024, 0, stream>>>(partials, out);
}
